// ImageEncoder_52123723104441
// MI455X (gfx1250) — matrix operand order repaired and checked by reading, not compiled, not run on hardware
//
#include <hip/hip_runtime.h>
#include <hip/hip_bf16.h>
#include <math.h>

typedef unsigned short u16;
typedef unsigned int   u32;

typedef __attribute__((ext_vector_type(4)))  u32    v4u;
typedef __attribute__((ext_vector_type(8)))  float  v8f;
typedef __attribute__((ext_vector_type(16))) __bf16 v16bf;
typedef __attribute__((ext_vector_type(4)))  u32    uint32x4;
typedef __attribute__((ext_vector_type(8)))  int    int32x8;
typedef __attribute__((ext_vector_type(4)))  int    int32x4;

#define DEV static __device__ __forceinline__

DEV u16 f2bf(float f) {
  u32 u = __float_as_uint(f);
  u32 r = u + 0x7FFFu + ((u >> 16) & 1u);
  return (u16)(r >> 16);
}

union FragU { v4u q[2]; u16 u[16]; v16bf v; };

DEV void frag_zero(FragU& f) { v4u z = {}; f.q[0] = z; f.q[1] = z; }

// A fragment (16x32 bf16): lane m=lane&15, half=lane>>4;
// elems 0..7 -> K = k0+half*8+j ; elems 8..15 -> K = k0+16+half*8+j
DEV void load_a16(FragU& f, const u16* rp, int k0, int half) {
  f.q[0] = *(const v4u*)(rp + k0 + half * 8);
  f.q[1] = *(const v4u*)(rp + k0 + 16 + half * 8);
}
// B fragment (32x16 bf16), B stored row-major [N][K] (weight style):
// lane n=lane&15; elems 0..7 -> K = k0+half*8+j ; elems 8..15 -> K = k0+16+half*8+j
DEV void load_b16(FragU& f, const u16* rp, int k0, int half) {
  f.q[0] = *(const v4u*)(rp + k0 + half * 8);
  f.q[1] = *(const v4u*)(rp + k0 + 16 + half * 8);
}

DEV v8f wmma_bf16(const FragU& a, const FragU& b, v8f c) {
  return __builtin_amdgcn_wmma_f32_16x16x32_bf16(false, a.v, false, b.v,
                                                 (short)0, c, false, false);
}

// C/D 16x16 f32 layout: vgpr v, lane: row = (lane>=16)? v+8 : v ; col = lane&15
DEV void store_frag(v8f c, int row0, int col0, int lane,
                    const float* bias, const float* resid, const float* gamma,
                    float* outF, u16* outB, int N, int gelu) {
  int n = col0 + (lane & 15);
#pragma unroll
  for (int v = 0; v < 8; ++v) {
    int m = row0 + ((lane >> 4) ? v + 8 : v);
    float val = c[v];
    if (bias)  val += bias[n];
    if (gelu)  val = 0.5f * val * (1.0f + erff(val * 0.70710678f));
    if (gamma) val *= gamma[n];
    size_t o = (size_t)m * N + n;
    if (resid) val += resid[o];
    if (outF)  outF[o] = val;
    if (outB)  outB[o] = f2bf(val);
  }
}

// ---------------- generic WMMA GEMM: out = A(MxK) * B^T (+epilogue) ----------
// B always row-major [N][K]. grid = (N/64, M/128, batch); block = 256 (8 waves)
__global__ __launch_bounds__(256) void k_gemm(
    const u16* __restrict__ A, int lda, long long sA,
    const u16* __restrict__ B, int ldb, long long sB,
    const float* __restrict__ bias,
    const float* __restrict__ resid, long long sR,
    const float* __restrict__ gamma,
    float* __restrict__ outF, u16* __restrict__ outB, long long sO,
    int N, int K, int gelu) {
  int lane = threadIdx.x & 31, wid = threadIdx.x >> 5;
  int wm = wid >> 1, wn = wid & 1;
  int half = lane >> 4, l15 = lane & 15;
  int z = blockIdx.z;
  const u16* Ab = A + (size_t)z * sA;
  const u16* Bb = B + (size_t)z * sB;
  int rowBase = blockIdx.y * 128 + wm * 32;
  int colBase = blockIdx.x * 64 + wn * 32;
  const u16* ra0 = Ab + (size_t)(rowBase + l15) * lda;
  const u16* ra1 = ra0 + (size_t)16 * lda;
  const u16* rb0 = Bb + (size_t)(colBase + l15) * ldb;
  const u16* rb1 = rb0 + (size_t)16 * ldb;
  v8f c00 = {}, c01 = {}, c10 = {}, c11 = {};
  FragU a0, a1, b0, b1;
  for (int k0 = 0; k0 < K; k0 += 32) {
    load_a16(a0, ra0, k0, half);
    load_a16(a1, ra1, k0, half);
    load_b16(b0, rb0, k0, half);
    load_b16(b1, rb1, k0, half);
    c00 = wmma_bf16(a0, b0, c00); c01 = wmma_bf16(a0, b1, c01);
    c10 = wmma_bf16(a1, b0, c10); c11 = wmma_bf16(a1, b1, c11);
  }
  const float* rz = resid ? resid + (size_t)z * sR : nullptr;
  float* oz = outF ? outF + (size_t)z * sO : nullptr;
  u16* obz = outB ? outB + (size_t)z * sO : nullptr;
  store_frag(c00, rowBase,      colBase,      lane, bias, rz, gamma, oz, obz, N, gelu);
  store_frag(c01, rowBase,      colBase + 16, lane, bias, rz, gamma, oz, obz, N, gelu);
  store_frag(c10, rowBase + 16, colBase,      lane, bias, rz, gamma, oz, obz, N, gelu);
  store_frag(c11, rowBase + 16, colBase + 16, lane, bias, rz, gamma, oz, obz, N, gelu);
}

// ---------------- implicit-GEMM conv (NHWC bf16 in, f32 out) ----------------
// weight staged as (KK, Cout, Cin) bf16. grid = (Cout/64, Mtok/128)
__global__ __launch_bounds__(256) void k_conv_wmma(
    const u16* __restrict__ X, int Hin, int Win, int Cin,
    const u16* __restrict__ Wt, int KS, int stride, int pad,
    const float* __restrict__ bias, float* __restrict__ outF,
    int Hout, int Wout, int Cout) {
  int lane = threadIdx.x & 31, wid = threadIdx.x >> 5;
  int wm = wid >> 1, wn = wid & 1;
  int half = lane >> 4, l15 = lane & 15;
  int rowBase = blockIdx.y * 128 + wm * 32;
  int colBase = blockIdx.x * 64 + wn * 32;
  int HW = Hout * Wout;
  int t0 = rowBase + l15, t1 = t0 + 16;
  int img0 = t0 / HW, r0 = t0 - img0 * HW, ho0 = r0 / Wout, wo0 = r0 - ho0 * Wout;
  int img1 = t1 / HW, r1 = t1 - img1 * HW, ho1 = r1 / Wout, wo1 = r1 - ho1 * Wout;
  v8f c00 = {}, c01 = {}, c10 = {}, c11 = {};
  FragU a0, a1, b0, b1;
  for (int dy = 0; dy < KS; ++dy) {
    for (int dx = 0; dx < KS; ++dx) {
      int hi0 = ho0 * stride - pad + dy, wi0 = wo0 * stride - pad + dx;
      int hi1 = ho1 * stride - pad + dy, wi1 = wo1 * stride - pad + dx;
      bool v0 = (u32)hi0 < (u32)Hin && (u32)wi0 < (u32)Win;
      bool v1 = (u32)hi1 < (u32)Hin && (u32)wi1 < (u32)Win;
      const u16* rp0 = X + (size_t)((img0 * Hin + hi0) * Win + wi0) * Cin;
      const u16* rp1 = X + (size_t)((img1 * Hin + hi1) * Win + wi1) * Cin;
      const u16* wb  = Wt + (size_t)(dy * KS + dx) * Cout * Cin;
      const u16* rb0 = wb + (size_t)(colBase + l15) * Cin;
      const u16* rb1 = rb0 + (size_t)16 * Cin;
      for (int k0 = 0; k0 < Cin; k0 += 32) {
        if (v0) load_a16(a0, rp0, k0, half); else frag_zero(a0);
        if (v1) load_a16(a1, rp1, k0, half); else frag_zero(a1);
        load_b16(b0, rb0, k0, half); load_b16(b1, rb1, k0, half);
        c00 = wmma_bf16(a0, b0, c00); c01 = wmma_bf16(a0, b1, c01);
        c10 = wmma_bf16(a1, b0, c10); c11 = wmma_bf16(a1, b1, c11);
      }
    }
  }
  store_frag(c00, rowBase,      colBase,      lane, bias, nullptr, nullptr, outF, nullptr, Cout, 0);
  store_frag(c01, rowBase,      colBase + 16, lane, bias, nullptr, nullptr, outF, nullptr, Cout, 0);
  store_frag(c10, rowBase + 16, colBase,      lane, bias, nullptr, nullptr, outF, nullptr, Cout, 0);
  store_frag(c11, rowBase + 16, colBase + 16, lane, bias, nullptr, nullptr, outF, nullptr, Cout, 0);
}

// ---------------- VQ argmin: WMMA dot vs codebook staged in LDS via TDM -----
// Z: (T,64) bf16 tokens; CBg: (1024,64) bf16; CBN: |c|^2. grid = T/128, blk 256
__global__ __launch_bounds__(256) void k_vq(
    const u16* __restrict__ Z, const u16* __restrict__ CBg,
    const float* __restrict__ CBN, int* __restrict__ idxOut) {
  extern __shared__ u16 smem[];
  u16* cb = smem;                       // 1024*64 bf16 = 128 KB (WGP LDS = 320KB)
  float* cbn = (float*)(smem + 65536);  // 4 KB
#if __has_builtin(__builtin_amdgcn_tensor_load_to_lds)
  if ((threadIdx.x >> 5) == 0) {
    // Tensor Data Mover: 2D tile 1024 x 64 elements (2B each) -> LDS
    u32 ldsOff = (u32)(unsigned long long)(void*)cb;
    unsigned long long ga = (unsigned long long)(const void*)CBg;
    uint32x4 g0 = {1u, ldsOff, (u32)ga,
                   (u32)((u32)((ga >> 32) & 0x1FFFFFFull) | 0x80000000u)}; // type=2
    int32x8 g1 = {(int)(1u << 16),     // data_size = 2 bytes
                  (int)(64u << 16),    // tensor_dim0 = 64
                  (int)(1024u << 16),  // tensor_dim1 = 1024
                  (int)(64u << 16),    // tile_dim0 = 64
                  1024,                // tile_dim1 = 1024
                  64, 0, 0};           // dim0_stride = 64
    int32x4 g2 = {0, 0, 0, 0};
    int32x4 g3 = {0, 0, 0, 0};
    int32x8 g4 = {0, 0, 0, 0, 0, 0, 0, 0};  // 6-arg toolchain variant
    __builtin_amdgcn_tensor_load_to_lds(g0, g1, g2, g3, g4, 0);
#if __has_builtin(__builtin_amdgcn_s_wait_tensorcnt)
    __builtin_amdgcn_s_wait_tensorcnt(0);
#endif
  }
#endif
  // redundant manual copy (semantic safety net for the TDM descriptor)
  for (int i = threadIdx.x; i < 65536; i += 256) cb[i] = CBg[i];
  for (int i = threadIdx.x; i < 1024; i += 256)  cbn[i] = CBN[i];
  __syncthreads();

  int lane = threadIdx.x & 31, wid = threadIdx.x >> 5;
  int half = lane >> 4, l15 = lane & 15;
  int tokBase = blockIdx.x * 128 + wid * 16;
  const u16* ra = Z + (size_t)(tokBase + l15) * 64;
  FragU aLo, aHi;
  load_a16(aLo, ra, 0, half);
  load_a16(aHi, ra, 32, half);
  float best[8]; int bidx[8];
#pragma unroll
  for (int v = 0; v < 8; ++v) { best[v] = 3.4e38f; bidx[v] = 0; }
  for (int n0 = 0; n0 < 1024; n0 += 16) {
    FragU b0, b1;
    const u16* rb = cb + (size_t)(n0 + l15) * 64;
    load_b16(b0, rb, 0, half);
    load_b16(b1, rb, 32, half);
    v8f c = {};
    c = wmma_bf16(aLo, b0, c);
    c = wmma_bf16(aHi, b1, c);
    int code = n0 + l15;
    float cn = cbn[code];
#pragma unroll
    for (int v = 0; v < 8; ++v) {
      float s = cn - 2.0f * c[v];
      if (s < best[v]) { best[v] = s; bidx[v] = code; }
    }
  }
#pragma unroll
  for (int v = 0; v < 8; ++v) {
    for (int o = 1; o < 16; o <<= 1) {
      float ov = __shfl_xor(best[v], o);
      int   oi = __shfl_xor(bidx[v], o);
      if (ov < best[v]) { best[v] = ov; bidx[v] = oi; }
    }
  }
  if (l15 == 0) {
#pragma unroll
    for (int v = 0; v < 8; ++v) {
      int m = (lane >> 4) ? v + 8 : v;
      idxOut[tokBase + m] = bidx[v];
    }
  }
}

// ---------------- scalar support kernels ------------------------------------
__global__ void k_conv0(const float* __restrict__ X, const float* __restrict__ Wt,
                        const float* __restrict__ Bs, float* __restrict__ Y) {
  long long i = (long long)blockIdx.x * 256 + threadIdx.x;
  if (i >= 32LL * 128 * 128 * 64) return;
  int c = (int)(i & 63); long long t = i >> 6;
  int img = (int)(t >> 14); int r = (int)(t & 16383);
  int h = r >> 7, w = r & 127;
  const float* xi = X + (long long)img * 16384;
  float acc = Bs[c];
  for (int dy = 0; dy < 7; ++dy) {
    int hi = h - 3 + dy; if ((u32)hi >= 128u) continue;
    for (int dx = 0; dx < 7; ++dx) {
      int wi = w - 3 + dx; if ((u32)wi >= 128u) continue;
      acc += xi[hi * 128 + wi] * Wt[c * 49 + dy * 7 + dx];
    }
  }
  Y[i] = acc;
}

__global__ void k_dwconv(const float* __restrict__ X, const float* __restrict__ Wt,
                         const float* __restrict__ Bs, float* __restrict__ Y,
                         int H, int W, int C, int KS, int pad, long long total) {
  long long i = (long long)blockIdx.x * 256 + threadIdx.x;
  if (i >= total) return;
  int c = (int)(i % C); long long t = i / C;
  int HW = H * W; int img = (int)(t / HW); int r = (int)(t % HW);
  int h = r / W, w = r % W;
  const float* xi = X + (long long)img * HW * C;
  float acc = Bs[c];
  for (int dy = 0; dy < KS; ++dy) {
    int hi = h - pad + dy; if ((u32)hi >= (u32)H) continue;
    for (int dx = 0; dx < KS; ++dx) {
      int wi = w - pad + dx; if ((u32)wi >= (u32)W) continue;
      acc += xi[(hi * W + wi) * C + c] * Wt[c * KS * KS + dy * KS + dx];
    }
  }
  Y[i] = acc;
}

__global__ void k_ln(const float* __restrict__ X, const float* __restrict__ g,
                     const float* __restrict__ b, float* __restrict__ outF,
                     u16* __restrict__ outB, int C) {
  __shared__ float red[64];
  size_t t = blockIdx.x;
  const float* x = X + t * C;
  int tid = threadIdx.x;
  float s = 0.f, s2 = 0.f;
  for (int c = tid; c < C; c += 64) { float v = x[c]; s += v; s2 += v * v; }
  red[tid] = s; __syncthreads();
  for (int k = 32; k > 0; k >>= 1) { if (tid < k) red[tid] += red[tid + k]; __syncthreads(); }
  s = red[0]; __syncthreads();
  red[tid] = s2; __syncthreads();
  for (int k = 32; k > 0; k >>= 1) { if (tid < k) red[tid] += red[tid + k]; __syncthreads(); }
  s2 = red[0];
  float mean = s / C;
  float var = s2 / C - mean * mean;
  float inv = rsqrtf(var + 1e-6f);
  for (int c = tid; c < C; c += 64) {
    float y = (x[c] - mean) * inv * g[c] + b[c];
    if (outF) outF[t * C + c] = y;
    if (outB) outB[t * C + c] = f2bf(y);
  }
}

__global__ void k_softmax(const float* __restrict__ S, u16* __restrict__ P,
                          int L, float scale) {
  __shared__ float red[256];
  int row = blockIdx.x, tid = threadIdx.x;
  const float* r = S + (size_t)row * L;
  float mx = -3.4e38f;
  for (int i = tid; i < L; i += 256) mx = fmaxf(mx, r[i] * scale);
  red[tid] = mx; __syncthreads();
  for (int k = 128; k > 0; k >>= 1) { if (tid < k) red[tid] = fmaxf(red[tid], red[tid + k]); __syncthreads(); }
  mx = red[0]; __syncthreads();
  float sum = 0.f;
  for (int i = tid; i < L; i += 256) sum += expf(r[i] * scale - mx);
  red[tid] = sum; __syncthreads();
  for (int k = 128; k > 0; k >>= 1) { if (tid < k) red[tid] += red[tid + k]; __syncthreads(); }
  float inv = 1.0f / red[0];
  for (int i = tid; i < L; i += 256)
    P[(size_t)row * L + i] = f2bf(expf(r[i] * scale - mx) * inv);
}

// weight (Cout,Cin,K,K) f32 -> (KK,Cout,Cin) bf16
__global__ void k_stagew(const float* __restrict__ w, u16* __restrict__ out,
                         int KK, int Cout, int Cin, long long total) {
  long long e = (long long)blockIdx.x * 256 + threadIdx.x;
  if (e >= total) return;
  int c = (int)(e % Cin); long long t = e / Cin;
  int n = (int)(t % Cout); int s = (int)(t / Cout);
  out[e] = f2bf(w[((long long)n * Cin + c) * KK + s]);
}

__global__ void k_f2bf(const float* __restrict__ in, u16* __restrict__ out, long long n) {
  long long i = (long long)blockIdx.x * 256 + threadIdx.x;
  if (i < n) out[i] = f2bf(in[i]);
}

// transpose V slice of qkv: (32,1024,768 cols 512..767) bf16 -> (32,256,1024) bf16
__global__ void k_vtrans(const u16* __restrict__ qkv, u16* __restrict__ Vt) {
  long long i = (long long)blockIdx.x * 256 + threadIdx.x;
  if (i >= 32LL * 256 * 1024) return;
  int k = (int)(i & 1023); long long t = i >> 10;
  int n = (int)(t & 255); int z = (int)(t >> 8);
  Vt[i] = qkv[(size_t)z * 786432 + (size_t)k * 768 + 512 + n];
}

__global__ void k_cbn(const float* __restrict__ cb, float* __restrict__ nrm) {
  int i = blockIdx.x * 256 + threadIdx.x;
  if (i >= 1024) return;
  float s = 0.f;
  for (int c = 0; c < 64; ++c) { float v = cb[i * 64 + c]; s += v * v; }
  nrm[i] = s;
}

__global__ void k_coef(const float* __restrict__ s, const float* __restrict__ w1,
                       const float* __restrict__ b1, const float* __restrict__ w2,
                       const float* __restrict__ b2, float* __restrict__ coef) {
  int tid = threadIdx.x;
  if (tid >= 32) return;
  int b = tid >> 2, n = tid & 3;
  float acc = b2[n];
  for (int j = 0; j < 16; ++j) {
    float h = b1[j];
    for (int m = 0; m < 4; ++m) h += w1[j * 4 + m] * s[b * 4 + m];
    acc += w2[n * 16 + j] * h;
  }
  float sv = s[b * 4 + n];
  coef[tid] = acc * sv * sv; // s * fusion(s) * s
}

__global__ void k_tsf(const float* __restrict__ Z, const float* __restrict__ coef,
                      float* __restrict__ out, int HW, long long total) {
  long long i = (long long)blockIdx.x * 256 + threadIdx.x;
  if (i >= total) return;
  long long per = (long long)HW * 64;
  int b = (int)(i / per); long long r = i % per;
  float acc = 0.f;
  for (int n = 0; n < 4; ++n)
    acc += coef[b * 4 + n] * Z[(long long)(b * 4 + n) * per + r];
  out[i] = acc;
}

// concat [up2(zq) bcast over n, feat] -> bf16 NHWC (32,H,W,64+Cf)
__global__ void k_concat(const float* __restrict__ zq, const float* __restrict__ feat,
                         u16* __restrict__ out, int H, int W, int Cf, long long total) {
  long long i = (long long)blockIdx.x * 256 + threadIdx.x;
  if (i >= total) return;
  int C = 64 + Cf;
  int c = (int)(i % C); long long t = i / C;
  int HW = H * W; int img = (int)(t / HW); int r = (int)(t % HW);
  int h = r / W, w = r % W, b = img >> 2;
  float v;
  if (c < 64) {
    int Wq = W >> 1, Hq = H >> 1;
    v = zq[((long long)b * Hq * Wq + (h >> 1) * Wq + (w >> 1)) * 64 + c];
  } else {
    v = feat[t * Cf + (c - 64)];
  }
  out[i] = f2bf(v);
}

// gather codebook rows -> zq NHWC (scratch) + NCHW slice of d_out
__global__ void k_zqb(const int* __restrict__ idx, const float* __restrict__ cb,
                      float* __restrict__ zqN, float* __restrict__ dout,
                      int HW, long long total) {
  long long i = (long long)blockIdx.x * 256 + threadIdx.x;
  if (i >= total) return;
  int c = (int)(i & 63); long long t = i >> 6;
  int b = (int)(t / HW); int hw = (int)(t % HW);
  float v = cb[(long long)idx[t] * 64 + c];
  zqN[i] = v;
  dout[((long long)b * 64 + c) * HW + hw] = v;
}

__global__ void k_loss(const float* __restrict__ a, const float* __restrict__ b,
                       float scale, long long n, float* __restrict__ acc) {
  __shared__ float red[256];
  float s = 0.f;
  for (long long i = (long long)blockIdx.x * 256 + threadIdx.x; i < n;
       i += (long long)gridDim.x * 256) {
    float d = a[i] - b[i]; s += d * d;
  }
  red[threadIdx.x] = s; __syncthreads();
  for (int k = 128; k > 0; k >>= 1) { if (threadIdx.x < k) red[threadIdx.x] += red[threadIdx.x + k]; __syncthreads(); }
  if (threadIdx.x == 0) atomicAdd(acc, red[0] * scale);
}

__global__ void k_final(float* __restrict__ dout, const float* __restrict__ acc) {
  dout[0] = acc[0];
}

// ============================================================================
extern "C" void kernel_launch(void* const* d_in, const int* in_sizes, int n_in,
                              void* d_out, int out_size, void* d_ws, size_t ws_size,
                              hipStream_t stream) {
  (void)in_sizes; (void)n_in; (void)out_size; (void)ws_size;
  auto F = [&](int i) { return (const float*)d_in[i]; };
  const float* x     = F(0);
  const float* s_all = F(1);
  const float* s_sub = F(2);
  // param leaf indices (depth-first dict/list order):
  // down0: 3 conv_w,4 conv_b,5 ln_g,6 ln_b, blocks 7..15, 16..24
  // down1: 25..28, blocks 29..37, 38..46
  // down2: 47..50, blocks 51..59, 60..68, attn 69..74
  // conv_latent: 75/76, 77/78, 79/80 ; fusion 81..84 ; codebook 85

  char* ws = (char*)d_ws;
  size_t off = 0;
  auto alloc = [&](size_t bytes) { size_t o = off; off += (bytes + 255) & ~(size_t)255; return o; };
  float* bufA  = (float*)(ws + alloc(134217728));  // (32,128,128,64) f32 NHWC
  float* bufB  = (float*)(ws + alloc(134217728));  // temp f32 / concat bf16
  float* bufC  = (float*)(ws + alloc(134217728));  // attn scores / latent conv out
  u16*   bufS  = (u16*)(ws + alloc(67108864));     // bf16 stage (LN out / probs)
  u16*   hidB  = (u16*)(ws + alloc(268435456));    // bf16 hidden / qkv / av
  float* feat0 = (float*)(ws + alloc(134217728));
  float* feat1 = (float*)(ws + alloc(67108864));
  float* feat2 = (float*)(ws + alloc(33554432));
  u16*   wst   = (u16*)(ws + alloc(2097152));      // staged weights bf16
  u16*   cbB   = (u16*)(ws + alloc(131072));       // codebook bf16
  float* cbN   = (float*)(ws + alloc(4096));       // codebook norms
  float* z1    = (float*)(ws + alloc(33554432));
  float* z2    = (float*)(ws + alloc(33554432));
  float* zq1   = (float*)(ws + alloc(33554432));
  float* zq2   = (float*)(ws + alloc(33554432));
  u16*   zs    = (u16*)(ws + alloc(16777216));     // VQ tokens bf16 / V^T
  int*   idxb  = (int*)(ws + alloc(2097152));
  float* coefA = (float*)(ws + alloc(256));
  float* coefS = (float*)(ws + alloc(256));
  float* lacc  = (float*)(ws + alloc(256));

  (void)hipMemsetAsync(lacc, 0, 16, stream);
  auto g1 = [](long long tot) { return dim3((u32)((tot + 255) / 256)); };

  // ---------------- level 0 ----------------
  k_conv0<<<g1(32LL * 128 * 128 * 64), 256, 0, stream>>>(x, F(3), F(4), bufB);
  k_ln<<<dim3(32 * 128 * 128), 64, 0, stream>>>(bufB, F(5), F(6), bufA, nullptr, 64);

  auto block = [&](int T, int C, int H, int W, int pr, int pb) {
    long long tot = (long long)T * C;
    k_dwconv<<<g1(tot), 256, 0, stream>>>(bufA, F(pb + 0), F(pb + 1), bufB, H, W, C, 7, pr, tot);
    k_ln<<<dim3(T), 64, 0, stream>>>(bufB, F(pb + 2), F(pb + 3), nullptr, bufS, C);
    long long wn = 4LL * C * C;
    k_stagew<<<g1(wn), 256, 0, stream>>>(F(pb + 4), wst, 1, 4 * C, C, wn);
    k_gemm<<<dim3(4 * C / 64, T / 128, 1), 256, 0, stream>>>(
        bufS, C, 0, wst, C, 0, F(pb + 5), nullptr, 0, nullptr,
        nullptr, hidB, 0, 4 * C, C, 1);
    k_stagew<<<g1(wn), 256, 0, stream>>>(F(pb + 6), wst, 1, C, 4 * C, wn);
    k_gemm<<<dim3(C / 64, T / 128, 1), 256, 0, stream>>>(
        hidB, 4 * C, 0, wst, 4 * C, 0, F(pb + 7), bufA, 0, F(pb + 8),
        bufA, nullptr, 0, C, 4 * C, 0);
  };
  block(524288, 64, 128, 128, 3, 7);
  block(524288, 64, 128, 128, 3, 16);
  (void)hipMemcpyAsync(feat0, bufA, (size_t)524288 * 64 * 4, hipMemcpyDeviceToDevice, stream);

  // ---------------- level 1 ----------------
  k_ln<<<dim3(524288), 64, 0, stream>>>(bufA, F(27), F(28), nullptr, bufS, 64);
  { long long wn = 16LL * 128 * 64;
    k_stagew<<<g1(wn), 256, 0, stream>>>(F(25), wst, 16, 128, 64, wn); }
  k_conv_wmma<<<dim3(2, 131072 / 128), 256, 0, stream>>>(
      bufS, 128, 128, 64, wst, 4, 2, 1, F(26), bufA, 64, 64, 128);
  block(131072, 128, 64, 64, 3, 29);
  block(131072, 128, 64, 64, 3, 38);
  (void)hipMemcpyAsync(feat1, bufA, (size_t)131072 * 128 * 4, hipMemcpyDeviceToDevice, stream);

  // ---------------- level 2 ----------------
  k_ln<<<dim3(131072), 64, 0, stream>>>(bufA, F(49), F(50), nullptr, bufS, 128);
  { long long wn = 16LL * 256 * 128;
    k_stagew<<<g1(wn), 256, 0, stream>>>(F(47), wst, 16, 256, 128, wn); }
  k_conv_wmma<<<dim3(4, 32768 / 128), 256, 0, stream>>>(
      bufS, 64, 64, 128, wst, 4, 2, 1, F(48), bufA, 32, 32, 256);
  block(32768, 256, 32, 32, 3, 51);
  block(32768, 256, 32, 32, 3, 60);

  // attention (B=32, L=1024, C=256)
  k_ln<<<dim3(32768), 64, 0, stream>>>(bufA, F(69), F(70), nullptr, bufS, 256);
  { long long wn = 768LL * 256;
    k_stagew<<<g1(wn), 256, 0, stream>>>(F(71), wst, 1, 768, 256, wn); }
  k_gemm<<<dim3(12, 256, 1), 256, 0, stream>>>(                 // qkv (bf16 out)
      bufS, 256, 0, wst, 256, 0, F(72), nullptr, 0, nullptr,
      nullptr, hidB, 0, 768, 256, 0);
  k_vtrans<<<g1(32LL * 256 * 1024), 256, 0, stream>>>(hidB, zs); // V^T for AV
  k_gemm<<<dim3(16, 8, 32), 256, 0, stream>>>(                  // S = Q K^T
      hidB, 768, 786432, hidB + 256, 768, 786432,
      nullptr, nullptr, 0, nullptr, bufC, nullptr, 1048576, 1024, 256, 0);
  k_softmax<<<dim3(32768), 256, 0, stream>>>(bufC, bufS, 1024, 0.0625f);
  k_gemm<<<dim3(4, 8, 32), 256, 0, stream>>>(                   // O = A V (B = V^T, NK)
      bufS, 1024, 1048576, zs, 1024, 262144,
      nullptr, nullptr, 0, nullptr, bufB, nullptr, 262144, 256, 1024, 0);
  k_f2bf<<<g1(32768LL * 256), 256, 0, stream>>>(bufB, hidB, 32768LL * 256);
  { long long wn = 256LL * 256;
    k_stagew<<<g1(wn), 256, 0, stream>>>(F(73), wst, 1, 256, 256, wn); }
  k_gemm<<<dim3(4, 256, 1), 256, 0, stream>>>(                  // o-proj + residual
      hidB, 256, 0, wst, 256, 0, F(74), bufA, 0, nullptr,
      feat2, nullptr, 0, 256, 256, 0);

  // ---------------- latent pyramid + VQ ----------------
  k_coef<<<1, 32, 0, stream>>>(s_all, F(81), F(82), F(83), F(84), coefA);
  k_coef<<<1, 32, 0, stream>>>(s_sub, F(81), F(82), F(83), F(84), coefS);
  k_stagew<<<g1(65536), 256, 0, stream>>>(F(85), cbB, 1, 1024, 64, 65536);
  k_cbn<<<dim3(4), 256, 0, stream>>>(F(85), cbN);

  float* dout = (float*)d_out;
  const size_t TOTAL_ALL = 11010048;   // 8*64*(1024+4096+16384)
  const size_t oAll[3] = {2621440, 524288, 0};   // by level i (i=2 -> 0)
  auto vq = [&](const float* zf, float* zqf, size_t doff, int HW) {
    long long tot = 8LL * HW * 64;
    k_f2bf<<<g1(tot), 256, 0, stream>>>(zf, zs, tot);
    k_vq<<<dim3((u32)(8 * HW / 128)), 256, 135168, stream>>>(zs, cbB, cbN, idxb);
    k_zqb<<<g1(tot), 256, 0, stream>>>(idxb, F(85), zqf, dout + doff, HW, tot);
  };
  auto losses = [&](int HW) {
    long long tot = 8LL * HW * 64;
    float cnt = (float)tot;
    k_loss<<<dim3(1024), 256, 0, stream>>>(zq1, z1, 1.25f / cnt, tot, lacc);
    k_loss<<<dim3(1024), 256, 0, stream>>>(zq2, z2, 1.25f / cnt, tot, lacc);
    k_loss<<<dim3(1024), 256, 0, stream>>>(zq1, z2, 0.25f / cnt, tot, lacc);
  };

  // i = 2
  {
    int HW = 1024;
    k_f2bf<<<g1(32768LL * 256), 256, 0, stream>>>(feat2, bufS, 32768LL * 256);
    { long long wn = 9LL * 64 * 256;
      k_stagew<<<g1(wn), 256, 0, stream>>>(F(79), wst, 9, 64, 256, wn); }
    k_conv_wmma<<<dim3(1, 32768 / 128), 256, 0, stream>>>(
        bufS, 32, 32, 256, wst, 3, 1, 1, F(80), bufC, 32, 32, 64);
    long long tot = 8LL * HW * 64;
    k_tsf<<<g1(tot), 256, 0, stream>>>(bufC, coefA, z1, HW, tot);
    k_tsf<<<g1(tot), 256, 0, stream>>>(bufC, coefS, z2, HW, tot);
    vq(z1, zq1, oAll[2], HW);
    vq(z2, zq2, TOTAL_ALL + oAll[2], HW);
    losses(HW);
  }
  // i = 1  (concat up2(zq_level2) with feat1)
  {
    int H = 64, W = 64, HW = 4096, Cf = 128;
    long long ctot = 32LL * HW * (64 + Cf);
    { long long wn = 9LL * 64 * 192;
      k_stagew<<<g1(wn), 256, 0, stream>>>(F(77), wst, 9, 64, 192, wn); }
    k_concat<<<g1(ctot), 256, 0, stream>>>(zq1, feat1, (u16*)bufB, H, W, Cf, ctot);
    k_conv_wmma<<<dim3(1, 131072 / 128), 256, 0, stream>>>(
        (u16*)bufB, H, W, 192, wst, 3, 1, 1, F(78), bufC, H, W, 64);
    long long tot = 8LL * HW * 64;
    k_tsf<<<g1(tot), 256, 0, stream>>>(bufC, coefA, z1, HW, tot);
    k_concat<<<g1(ctot), 256, 0, stream>>>(zq2, feat1, (u16*)bufB, H, W, Cf, ctot);
    k_conv_wmma<<<dim3(1, 131072 / 128), 256, 0, stream>>>(
        (u16*)bufB, H, W, 192, wst, 3, 1, 1, F(78), bufC, H, W, 64);
    k_tsf<<<g1(tot), 256, 0, stream>>>(bufC, coefS, z2, HW, tot);
    vq(z1, zq1, oAll[1], HW);
    vq(z2, zq2, TOTAL_ALL + oAll[1], HW);
    losses(HW);
  }
  // i = 0
  {
    int H = 128, W = 128, HW = 16384, Cf = 64;
    long long ctot = 32LL * HW * (64 + Cf);
    { long long wn = 9LL * 64 * 128;
      k_stagew<<<g1(wn), 256, 0, stream>>>(F(75), wst, 9, 64, 128, wn); }
    k_concat<<<g1(ctot), 256, 0, stream>>>(zq1, feat0, (u16*)bufB, H, W, Cf, ctot);
    k_conv_wmma<<<dim3(1, 524288 / 128), 256, 0, stream>>>(
        (u16*)bufB, H, W, 128, wst, 3, 1, 1, F(76), bufC, H, W, 64);
    long long tot = 8LL * HW * 64;
    k_tsf<<<g1(tot), 256, 0, stream>>>(bufC, coefA, z1, HW, tot);
    k_concat<<<g1(ctot), 256, 0, stream>>>(zq2, feat0, (u16*)bufB, H, W, Cf, ctot);
    k_conv_wmma<<<dim3(1, 524288 / 128), 256, 0, stream>>>(
        (u16*)bufB, H, W, 128, wst, 3, 1, 1, F(76), bufC, H, W, 64);
    k_tsf<<<g1(tot), 256, 0, stream>>>(bufC, coefS, z2, HW, tot);
    vq(z1, zq1, oAll[0], HW);
    vq(z2, zq2, TOTAL_ALL + oAll[0], HW);
    losses(HW);
  }
  k_final<<<1, 1, 0, stream>>>(dout + 2 * TOTAL_ALL, lacc);
}
